// CommCellInnerRNN_12695923326983
// MI455X (gfx1250) — matrix operand order repaired and checked by reading, not compiled, not run on hardware
//
#include <hip/hip_runtime.h>
#include <hip/hip_bf16.h>

// ---------------------------------------------------------------------------
// Types for CDNA5 WMMA
// ---------------------------------------------------------------------------
typedef __attribute__((ext_vector_type(16))) __bf16 v16bf;
typedef __attribute__((ext_vector_type(8)))  float  v8f;

static __device__ __forceinline__ __bf16 f2bf(float x) {
  // round-to-nearest-even fp32 -> bf16 (inputs are finite)
  unsigned u = __builtin_bit_cast(unsigned, x);
  u += 0x7FFFu + ((u >> 16) & 1u);
  unsigned short h = (unsigned short)(u >> 16);
  return __builtin_bit_cast(__bf16, h);
}
static __device__ __forceinline__ float sigf(float x) {
  return 1.0f / (1.0f + __expf(-x));
}

// ---------------------------------------------------------------------------
// GEMM: C[M x N] = A[M x Kpad] * B[Kpad x N], with B given transposed
// (Bt is N x Kpad, row-major, bf16, Kpad % 32 == 0, zero-padded).
// One wave computes one 16x16 tile via v_wmma_f32_16x16x32_bf16.
// Block = 256 threads = 8 waves covering a 16(M) x 128(N) tile.
// grid = (M/16, N/128)
// ---------------------------------------------------------------------------
__global__ __launch_bounds__(256) void wmma_gemm_bf16(
    const __bf16* __restrict__ A, const __bf16* __restrict__ Bt,
    float* __restrict__ C, int N, int Kpad)
{
  const int lane = threadIdx.x & 31;
  const int wave = threadIdx.x >> 5;
  const int r    = lane & 15;
  const int hi   = lane >> 4;                       // lane half
  const long tileM = (long)blockIdx.x * 16;
  const long tileN = (long)blockIdx.y * 128 + wave * 16;

  // A fragment (16x32 bf16): lanes 0-15 row M=r hold K {0..7,16..23};
  // lanes 16-31 row M=r hold K {8..15,24..31}  (ISA 7.12.2)
  const __bf16* aPtr = A  + (tileM + r) * (long)Kpad + hi * 8;
  // B fragment (32x16 bf16): lanes 0-15 col N=r hold K {0..7,16..23};
  // lanes 16-31 col N=r hold K {8..15,24..31}
  const __bf16* bPtr = Bt + (tileN + r) * (long)Kpad + hi * 8;

  v8f acc = {};
  for (int k = 0; k < Kpad; k += 32) {
    v16bf a, b;
    ((int4*)&a)[0] = *(const int4*)(aPtr + k);
    ((int4*)&a)[1] = *(const int4*)(aPtr + k + 16);
    ((int4*)&b)[0] = *(const int4*)(bPtr + k);
    ((int4*)&b)[1] = *(const int4*)(bPtr + k + 16);
    __builtin_prefetch(aPtr + k + 128, 0, 1);
    __builtin_prefetch(bPtr + k + 128, 0, 1);
    acc = __builtin_amdgcn_wmma_f32_16x16x32_bf16(
        false, a, false, b, (short)0, acc, false, false);
  }
  // C/D layout: lane half -> M base, VGPR v -> row offset, lane&15 -> N
  float* cPtr = C + (tileM + hi * 8) * (long)N + tileN + r;
#pragma unroll
  for (int v = 0; v < 8; ++v) cPtr[(long)v * N] = acc[v];
}

// ---------------------------------------------------------------------------
// Weight prep: dst[n*ldDst + koff + k] = bf16(src[k*Nc + n])   (transpose)
// ---------------------------------------------------------------------------
__global__ void transpose_f32_to_bf16(__bf16* __restrict__ dst, int ldDst, int koff,
                                      const float* __restrict__ src, int K, int Nc)
{
  long idx = (long)blockIdx.x * blockDim.x + threadIdx.x;
  long total = (long)K * Nc;
  if (idx >= total) return;
  int k = (int)(idx / Nc), n = (int)(idx % Nc);
  dst[(long)n * ldDst + koff + k] = f2bf(src[idx]);
}

__global__ void copy_f32_to_bf16(__bf16* __restrict__ dst, int ldDst, int coff,
                                 const float* __restrict__ src, int cols, long total)
{
  long idx = (long)blockIdx.x * blockDim.x + threadIdx.x;
  if (idx >= total) return;
  int r = (int)(idx / cols), c = (int)(idx % cols);
  dst[(long)r * ldDst + coff + c] = f2bf(src[idx]);
}

__global__ void copy_f32(float* __restrict__ dst, const float* __restrict__ src, long n)
{
  long idx = (long)blockIdx.x * blockDim.x + threadIdx.x;
  if (idx < n) dst[idx] = src[idx];
}

__global__ void fill_bf16_zero(__bf16* __restrict__ dst, long n)
{
  long idx = (long)blockIdx.x * blockDim.x + threadIdx.x;
  if (idx < n) dst[idx] = __builtin_bit_cast(__bf16, (unsigned short)0);
}

// zero out_signals region of d_out (cols 0..511 of 1024-wide rows)
__global__ void zero_out_sig(float* __restrict__ out, long total)
{
  long idx = (long)blockIdx.x * blockDim.x + threadIdx.x;
  if (idx >= total) return;
  int r = (int)(idx / 512), c = (int)(idx % 512);
  out[(long)r * 1024 + c] = 0.0f;
}

// ---------------------------------------------------------------------------
// Plain LSTM pointwise (layers 1/2): gates Z (M x 4U), bias, c_prev fp32.
// Optionally writes h as fp32 (strided) and always as bf16 (strided).
// ---------------------------------------------------------------------------
__global__ void lstm_point(const float* __restrict__ Z, const float* __restrict__ b,
                           const float* __restrict__ cprev, int U,
                           float* __restrict__ hF32, int ldF, int coffF,
                           __bf16* __restrict__ hBf, int ldB, int coffB, long total)
{
  long idx = (long)blockIdx.x * blockDim.x + threadIdx.x;
  if (idx >= total) return;
  int r = (int)(idx / U), u = (int)(idx % U);
  const float* z = Z + (long)r * 4 * U;
  float zi = z[u]         + b[u];
  float zf = z[U + u]     + b[U + u];
  float zg = z[2 * U + u] + b[2 * U + u];
  float zo = z[3 * U + u] + b[3 * U + u];
  float c = sigf(zf) * cprev[idx] + sigf(zi) * tanhf(zg);
  float h = sigf(zo) * tanhf(c);
  if (hF32) hF32[(long)r * ldF + coffF + u] = h;
  hBf[(long)r * ldB + coffB + u] = f2bf(h);
}

// ---------------------------------------------------------------------------
// Comm-step feature builder: Acomm row = [gathered(256)|onehot(4)|dist(1)|...]
// grid = 4096 blocks x 256 threads, t in [0,16]
// ---------------------------------------------------------------------------
__global__ void build_xt(__bf16* __restrict__ Acomm, const int* __restrict__ pidx,
                         const float* __restrict__ signals, int t)
{
  int r = blockIdx.x;
  int c = threadIdx.x;
  int pres, dist = 0, dir = -1;
  if (t < 16) {
    const int* p = pidx + (long)r * 48 + 3 * t;
    pres = p[0]; dist = p[1]; dir = p[2];
  } else {
    pres = r;                       // appended self column
  }
  float g = (pres >= 0) ? signals[(long)pres * 256 + c] : 0.0f;
  __bf16* row = Acomm + (long)r * 1312;
  row[c] = f2bf(g);
  if (c < 4)       row[256 + c] = f2bf((c == dir) ? 1.0f : 0.0f);
  else if (c == 4) row[260]     = f2bf((float)dist);
}

// ---------------------------------------------------------------------------
// Comm LSTM1 pointwise: U=512. nh1 (unmasked) -> Acomm2[:,0:512] as bf16;
// masked carry: cc1 fp32 in place, h-carry -> Acomm[:,773:1285] bf16.
// ---------------------------------------------------------------------------
__global__ void comm1_point(const float* __restrict__ Z, const float* __restrict__ b,
                            const int* __restrict__ pidx, int t,
                            float* __restrict__ cc1,
                            __bf16* __restrict__ Acomm, __bf16* __restrict__ Acomm2,
                            long total)
{
  long idx = (long)blockIdx.x * blockDim.x + threadIdx.x;
  if (idx >= total) return;
  int r = (int)(idx / 512), u = (int)(idx % 512);
  bool m = (t < 16) ? (pidx[(long)r * 48 + 3 * t] >= 0) : true;
  const float* z = Z + (long)r * 2048;
  float zi = z[u]        + b[u];
  float zf = z[512 + u]  + b[512 + u];
  float zg = z[1024 + u] + b[1024 + u];
  float zo = z[1536 + u] + b[1536 + u];
  float cnew = sigf(zf) * cc1[idx] + sigf(zi) * tanhf(zg);
  float h = sigf(zo) * tanhf(cnew);
  Acomm2[(long)r * 768 + u] = f2bf(h);            // LSTM2 input this step (unmasked)
  if (m) {
    cc1[idx] = cnew;
    Acomm[(long)r * 1312 + 773 + u] = f2bf(h);    // next step's h-carry
  }
}

// ---------------------------------------------------------------------------
// Comm LSTM2 pointwise: U=256. Masked: cc2 fp32, h-carry -> Acomm2[:,512:768],
// out_signals -> d_out cols [0,256) and [256,512).
// ---------------------------------------------------------------------------
__global__ void comm2_point(const float* __restrict__ Z, const float* __restrict__ b,
                            const int* __restrict__ pidx, int t,
                            float* __restrict__ cc2,
                            __bf16* __restrict__ Acomm2, float* __restrict__ out,
                            long total)
{
  long idx = (long)blockIdx.x * blockDim.x + threadIdx.x;
  if (idx >= total) return;
  int r = (int)(idx / 256), u = (int)(idx % 256);
  bool m = (t < 16) ? (pidx[(long)r * 48 + 3 * t] >= 0) : true;
  const float* z = Z + (long)r * 1024;
  float zi = z[u]       + b[u];
  float zf = z[256 + u] + b[256 + u];
  float zg = z[512 + u] + b[512 + u];
  float zo = z[768 + u] + b[768 + u];
  float cnew = sigf(zf) * cc2[idx] + sigf(zi) * tanhf(zg);
  float h = sigf(zo) * tanhf(cnew);
  if (m) {
    cc2[idx] = cnew;
    Acomm2[(long)r * 768 + 512 + u] = f2bf(h);    // next step's h-carry
    out[(long)r * 1024 + u]       = h;            // out_signals
    out[(long)r * 1024 + 256 + u] = h;            // duplicated copy
  }
}

// ---------------------------------------------------------------------------
// Host orchestration
// ---------------------------------------------------------------------------
extern "C" void kernel_launch(void* const* d_in, const int* in_sizes, int n_in,
                              void* d_out, int out_size, void* d_ws, size_t ws_size,
                              hipStream_t stream)
{
  (void)in_sizes; (void)n_in; (void)out_size; (void)ws_size;
  const int N = 4096;

  const float* inputs  = (const float*)d_in[0];
  const int*   pidx    = (const int*)  d_in[1];
  const float* rnn_h1  = (const float*)d_in[2];
  const float* rnn_c1  = (const float*)d_in[3];
  const float* rnn_h2  = (const float*)d_in[4];
  const float* rnn_c2  = (const float*)d_in[5];
  const float* comm_h1 = (const float*)d_in[6];
  const float* comm_c1 = (const float*)d_in[7];
  const float* comm_h2 = (const float*)d_in[8];
  const float* comm_c2 = (const float*)d_in[9];
  const float* signals = (const float*)d_in[10];
  const float* Wx1  = (const float*)d_in[11];
  const float* Wh1  = (const float*)d_in[12];
  const float* b1   = (const float*)d_in[13];
  const float* Wx2  = (const float*)d_in[14];
  const float* Wh2  = (const float*)d_in[15];
  const float* b2   = (const float*)d_in[16];
  const float* CWx1 = (const float*)d_in[17];
  const float* CWh1 = (const float*)d_in[18];
  const float* Cb1  = (const float*)d_in[19];
  const float* CWx2 = (const float*)d_in[20];
  const float* CWh2 = (const float*)d_in[21];
  const float* Cb2  = (const float*)d_in[22];
  float* out = (float*)d_out;

  // workspace carve-up
  char* ws = (char*)d_ws;
  auto carve = [&](size_t bytes) {
    char* p = ws; ws += (bytes + 255) & ~(size_t)255; return p;
  };
  __bf16* W1t    = (__bf16*)carve(2048ULL * 1024 * 2);  // [Wx1;Wh1]^T  (2048 x 1024)
  __bf16* W2t    = (__bf16*)carve(2048ULL * 1024 * 2);  // [Wx2;Wh2]^T
  __bf16* CW1t   = (__bf16*)carve(2048ULL * 1312 * 2);  // [CWx1;CWh1]^T, K padded 1285->1312
  __bf16* CW2t   = (__bf16*)carve(1024ULL * 768  * 2);  // [CWx2;CWh2]^T
  __bf16* A1     = (__bf16*)carve((size_t)N * 1024 * 2); // [inputs | rnn_h1]
  __bf16* A2     = (__bf16*)carve((size_t)N * 1024 * 2); // [h1 | rnn_h2]
  __bf16* Acomm  = (__bf16*)carve((size_t)N * 1312 * 2); // [xt(773) | ch1(512) | pad]
  __bf16* Acomm2 = (__bf16*)carve((size_t)N * 768  * 2); // [nh1(512) | ch2(256)]
  float*  Z      = (float*) carve((size_t)N * 2048 * 4); // gates scratch (reused)
  float*  cc1    = (float*) carve((size_t)N * 512  * 4);
  float*  cc2    = (float*) carve((size_t)N * 256  * 4);

  auto g1 = [](long total) { return dim3((unsigned)((total + 255) / 256)); };
  const dim3 B256(256);

  // --- init: zero padded buffers, zero out_signals region of d_out ---------
  fill_bf16_zero<<<g1(2048L * 1312), B256, 0, stream>>>(CW1t, 2048L * 1312);
  fill_bf16_zero<<<g1((long)N * 1312), B256, 0, stream>>>(Acomm, (long)N * 1312);
  zero_out_sig<<<g1((long)N * 512), B256, 0, stream>>>(out, (long)N * 512);

  // --- weight prep (transpose + bf16, stacked [Wx;Wh]) ---------------------
  transpose_f32_to_bf16<<<g1(512L * 2048), B256, 0, stream>>>(W1t, 1024, 0,   Wx1, 512, 2048);
  transpose_f32_to_bf16<<<g1(512L * 2048), B256, 0, stream>>>(W1t, 1024, 512, Wh1, 512, 2048);
  transpose_f32_to_bf16<<<g1(512L * 2048), B256, 0, stream>>>(W2t, 1024, 0,   Wx2, 512, 2048);
  transpose_f32_to_bf16<<<g1(512L * 2048), B256, 0, stream>>>(W2t, 1024, 512, Wh2, 512, 2048);
  transpose_f32_to_bf16<<<g1(773L * 2048), B256, 0, stream>>>(CW1t, 1312, 0,   CWx1, 773, 2048);
  transpose_f32_to_bf16<<<g1(512L * 2048), B256, 0, stream>>>(CW1t, 1312, 773, CWh1, 512, 2048);
  transpose_f32_to_bf16<<<g1(512L * 1024), B256, 0, stream>>>(CW2t, 768, 0,   CWx2, 512, 1024);
  transpose_f32_to_bf16<<<g1(256L * 1024), B256, 0, stream>>>(CW2t, 768, 512, CWh2, 256, 1024);

  // --- activation init -----------------------------------------------------
  copy_f32_to_bf16<<<g1((long)N * 512), B256, 0, stream>>>(A1, 1024, 0,   inputs,  512, (long)N * 512);
  copy_f32_to_bf16<<<g1((long)N * 512), B256, 0, stream>>>(A1, 1024, 512, rnn_h1,  512, (long)N * 512);
  copy_f32_to_bf16<<<g1((long)N * 512), B256, 0, stream>>>(A2, 1024, 512, rnn_h2,  512, (long)N * 512);
  copy_f32_to_bf16<<<g1((long)N * 512), B256, 0, stream>>>(Acomm,  1312, 773, comm_h1, 512, (long)N * 512);
  copy_f32_to_bf16<<<g1((long)N * 256), B256, 0, stream>>>(Acomm2, 768,  512, comm_h2, 256, (long)N * 256);
  copy_f32<<<g1((long)N * 512), B256, 0, stream>>>(cc1, comm_c1, (long)N * 512);
  copy_f32<<<g1((long)N * 256), B256, 0, stream>>>(cc2, comm_c2, (long)N * 256);

  // --- RNN layer 1: gates = [x|h] @ [Wx;Wh] --------------------------------
  wmma_gemm_bf16<<<dim3(N / 16, 2048 / 128), B256, 0, stream>>>(A1, W1t, Z, 2048, 1024);
  lstm_point<<<g1((long)N * 512), B256, 0, stream>>>(
      Z, b1, rnn_c1, 512, (float*)nullptr, 0, 0, A2, 1024, 0, (long)N * 512);

  // --- RNN layer 2 (features) ----------------------------------------------
  wmma_gemm_bf16<<<dim3(N / 16, 2048 / 128), B256, 0, stream>>>(A2, W2t, Z, 2048, 1024);
  // fp32 features -> d_out cols [512,1024); bf16 features -> Acomm cols [261,773)
  lstm_point<<<g1((long)N * 512), B256, 0, stream>>>(
      Z, b2, rnn_c2, 512, out, 1024, 512, Acomm, 1312, 261, (long)N * 512);

  // --- comm loop: 17 sequential steps --------------------------------------
  for (int t = 0; t <= 16; ++t) {
    build_xt<<<dim3(N), B256, 0, stream>>>(Acomm, pidx, signals, t);
    wmma_gemm_bf16<<<dim3(N / 16, 2048 / 128), B256, 0, stream>>>(Acomm, CW1t, Z, 2048, 1312);
    comm1_point<<<g1((long)N * 512), B256, 0, stream>>>(
        Z, Cb1, pidx, t, cc1, Acomm, Acomm2, (long)N * 512);
    wmma_gemm_bf16<<<dim3(N / 16, 1024 / 128), B256, 0, stream>>>(Acomm2, CW2t, Z, 1024, 768);
    comm2_point<<<g1((long)N * 256), B256, 0, stream>>>(
        Z, Cb2, pidx, t, cc2, Acomm2, out, (long)N * 256);
  }
}
